// Model_39298950758757
// MI455X (gfx1250) — matrix operand order repaired and checked by reading, not compiled, not run on hardware
//
#include <hip/hip_runtime.h>
#include <hip/hip_bf16.h>

// ---------------------------------------------------------------------------
// Types for CDNA5 WMMA
// ---------------------------------------------------------------------------
typedef __attribute__((ext_vector_type(16))) __bf16 v16bf;
typedef __attribute__((ext_vector_type(8)))  float  v8f;

union FragBF { uint4 q[2]; v16bf v; };

__device__ __forceinline__ unsigned short f32_to_bf16(float f) {
    unsigned u = __builtin_bit_cast(unsigned, f);
    unsigned r = u + 0x7FFFu + ((u >> 16) & 1u);   // round-to-nearest-even
    return (unsigned short)(r >> 16);
}

// ---------------------------------------------------------------------------
// CDNA5 async global->LDS copy (GLOBAL_LOAD_ASYNC_TO_LDS_B128, ASYNCcnt).
// Builtin signature (probe-confirmed by the round-2 diagnostic):
//   (int4 addrspace(1)* src, int4 addrspace(3)* dst, Ii offset, Ii cpol)
// where int4 is the GCC vector_size(16) spelling. We reach the exact types
// via integer casts: AS1 is numerically the generic global address; the AS3
// offset is the low 32 bits of the generic LDS address (ISA aperture rule).
// ---------------------------------------------------------------------------
typedef int v4i_vs __attribute__((vector_size(16)));
typedef __attribute__((address_space(1))) v4i_vs* as1_v4i;
typedef __attribute__((address_space(3))) v4i_vs* as3_v4i;

__device__ __forceinline__ void async_copy_b128(const void* gsrc, void* ldst) {
#if defined(__has_builtin) && __has_builtin(__builtin_amdgcn_global_load_async_to_lds_b128)
    __builtin_amdgcn_global_load_async_to_lds_b128(
        (as1_v4i)(unsigned long long)gsrc,
        (as3_v4i)(unsigned long long)(unsigned)(unsigned long long)ldst,
        0, 0);
#else
    unsigned lds = (unsigned)(unsigned long long)ldst;
    asm volatile("global_load_async_to_lds_b128 %0, %1, off"
                 :: "v"(lds), "v"(gsrc) : "memory");
#endif
}

__device__ __forceinline__ void wait_async0() {
#if defined(__has_builtin) && __has_builtin(__builtin_amdgcn_s_wait_asynccnt)
    __builtin_amdgcn_s_wait_asynccnt(0);
#else
    asm volatile("s_wait_asynccnt 0" ::: "memory");
#endif
}

// ---------------------------------------------------------------------------
// Tiled bf16 GEMM with f32 accumulation on v_wmma_f32_16x16x32_bf16.
//   C[M,N] = A[M,K] * B + bias (+ReLU)
//   TRANSB=true : B is stored [N,K] row-major (N-major) -> async direct fill
//   TRANSB=false: B is stored [K,N] row-major -> transposing LDS fill
// Block = 256 threads (8 waves). Tile 128x128x32. Wave tile 32x64 (2x4 WMMA).
// Batched via blockIdx.z with element strides. causal!=0 skips blocks fully
// above the diagonal (used for attention scores).
// M,N multiples of 128; K multiple of 32 (all shapes here satisfy this).
// ---------------------------------------------------------------------------
#define BM 128
#define BN 128
#define BK 32
#define SA_LD 40   // row stride in halves (80B) -> bank-conflict padding
#define SB_LD 40

template<bool TRANSB, bool RELU>
__global__ __launch_bounds__(256) void gemm_bf16_kernel(
    const unsigned short* __restrict__ A, long long sAb, int lda,
    const unsigned short* __restrict__ B, long long sBb, int ldb,
    const float* __restrict__ bias,
    float* __restrict__ Cf, long long sCfb,
    unsigned short* __restrict__ Ch, long long sChb,
    int M, int N, int K, int ldc, int causal)
{
    __shared__ alignas(16) unsigned short sA[BM * SA_LD];
    __shared__ alignas(16) unsigned short sB[BN * SB_LD];

    const int m0 = blockIdx.y * BM;
    const int n0 = blockIdx.x * BN;
    if (causal && n0 > m0 + (BM - 1)) return;   // fully-masked score block

    const long long z = blockIdx.z;
    A += z * sAb;
    B += z * sBb;

    const int tid  = threadIdx.x;
    const int lane = tid & 31;
    const int wave = tid >> 5;     // 0..7
    const int wm   = wave >> 1;    // 0..3 along M
    const int wn   = wave & 1;     // 0..1 along N
    const int l16  = lane & 15;
    const int lh   = lane >> 4;    // half-wave id

    v8f zero = {};
    v8f acc[2][4];
#pragma unroll
    for (int i = 0; i < 2; ++i)
#pragma unroll
        for (int j = 0; j < 4; ++j) acc[i][j] = zero;

    for (int k0 = 0; k0 < K; k0 += BK) {
        __syncthreads();   // previous iteration's fragment reads complete
        // ---- stage A tile (128x32 bf16, 8KB): 2 async 16B copies/thread ----
#pragma unroll
        for (int i = 0; i < 2; ++i) {
            int chunk = tid + i * 256;          // 0..511
            int r = chunk >> 2;                 // 0..127
            int c = (chunk & 3) << 3;           // 0,8,16,24
            async_copy_b128(A + (long long)(m0 + r) * lda + k0 + c,
                            &sA[r * SA_LD + c]);
        }
        // ---- stage B tile as [n][k] ----
        if (TRANSB) {
#pragma unroll
            for (int i = 0; i < 2; ++i) {
                int chunk = tid + i * 256;
                int r = chunk >> 2;
                int c = (chunk & 3) << 3;
                async_copy_b128(B + (long long)(n0 + r) * ldb + k0 + c,
                                &sB[r * SB_LD + c]);
            }
        } else {
#pragma unroll
            for (int i = 0; i < 2; ++i) {
                int chunk = tid + i * 256;
                int kk = chunk >> 4;            // 0..31
                int nn = (chunk & 15) << 3;     // 0..120
                uint4 d = *reinterpret_cast<const uint4*>(
                    B + (long long)(k0 + kk) * ldb + n0 + nn);
                const unsigned short* ds =
                    reinterpret_cast<const unsigned short*>(&d);
#pragma unroll
                for (int j = 0; j < 8; ++j)
                    sB[(nn + j) * SB_LD + kk] = ds[j];
            }
        }
        wait_async0();     // this wave's async copies have landed in LDS
        __syncthreads();   // all waves' tiles are visible

        // ---- load WMMA fragments per ISA 16-bit layouts ----
        FragBF a[2], b[4];
#pragma unroll
        for (int mt = 0; mt < 2; ++mt) {
            int m = wm * 32 + mt * 16 + l16;
            // A 16x32: lanes 0-15 K=0..7 & 16..23 ; lanes 16-31 K=8..15 & 24..31
            a[mt].q[0] = *reinterpret_cast<const uint4*>(&sA[m * SA_LD + lh * 8]);
            a[mt].q[1] = *reinterpret_cast<const uint4*>(&sA[m * SA_LD + 16 + lh * 8]);
        }
#pragma unroll
        for (int nt = 0; nt < 4; ++nt) {
            int n = wn * 64 + nt * 16 + l16;
            // B 32x16: lanes 0-15 K=0..7 & 16..23 ; lanes 16-31 K=8..15 & 24..31
            b[nt].q[0] = *reinterpret_cast<const uint4*>(&sB[n * SB_LD + lh * 8]);
            b[nt].q[1] = *reinterpret_cast<const uint4*>(&sB[n * SB_LD + 16 + lh * 8]);
        }
#pragma unroll
        for (int mt = 0; mt < 2; ++mt)
#pragma unroll
            for (int nt = 0; nt < 4; ++nt)
                acc[mt][nt] = __builtin_amdgcn_wmma_f32_16x16x32_bf16(
                    false, a[mt].v, false, b[nt].v,
                    (short)0, acc[mt][nt], false, false);
    }

    // ---- epilogue: bias / relu / store f32 and-or bf16 ----
    float* cf = Cf ? Cf + z * sCfb : nullptr;
    unsigned short* ch = Ch ? Ch + z * sChb : nullptr;
#pragma unroll
    for (int mt = 0; mt < 2; ++mt) {
#pragma unroll
        for (int nt = 0; nt < 4; ++nt) {
            int gn = n0 + wn * 64 + nt * 16 + l16;
            float bv = bias ? bias[gn] : 0.0f;
#pragma unroll
            for (int r = 0; r < 8; ++r) {
                int gm = m0 + wm * 32 + mt * 16 + lh * 8 + r;
                float v = acc[mt][nt][r] + bv;
                if (RELU) v = fmaxf(v, 0.0f);
                long long off = (long long)gm * ldc + gn;
                if (cf) cf[off] = v;
                if (ch) ch[off] = f32_to_bf16(v);
            }
        }
    }
}

// ---------------------------------------------------------------------------
// Weight convert + transpose: f32 [L][K][N] -> bf16 [L][N][K]  (N-major)
// ---------------------------------------------------------------------------
__global__ __launch_bounds__(256) void convT_kernel(
    const float* __restrict__ W, unsigned short* __restrict__ Wt, int K, int N)
{
    long long l = blockIdx.y;
    const float* w = W + l * (long long)K * N;
    unsigned short* o = Wt + l * (long long)N * K;
    long long total = (long long)K * N;
    for (long long i = (long long)blockIdx.x * 256 + threadIdx.x; i < total;
         i += (long long)gridDim.x * 256) {
        long long n = i / K, k = i % K;
        o[i] = f32_to_bf16(w[k * (long long)N + n]);
    }
}

// ---------------------------------------------------------------------------
// Embedding gather: h[row] = emb[x[row]]; mirrors f32 + bf16
// ---------------------------------------------------------------------------
__global__ __launch_bounds__(256) void embed_kernel(
    const int* __restrict__ x, const float* __restrict__ emb,
    float* __restrict__ h, unsigned short* __restrict__ hbf, int D)
{
    long long row = blockIdx.x;
    const float* e = emb + (long long)x[row] * D;
    for (int d = threadIdx.x; d < D; d += 256) {
        float v = e[d];
        h[row * D + d]   = v;
        hbf[row * D + d] = f32_to_bf16(v);
    }
}

// ---------------------------------------------------------------------------
// Row-wise causal softmax; scores f32 [B][S][S] -> P bf16 (zeros above diag)
// ---------------------------------------------------------------------------
__global__ __launch_bounds__(256) void softmax_kernel(
    const float* __restrict__ Sc, unsigned short* __restrict__ P, int S)
{
    __shared__ float red[256];
    int m = blockIdx.x;
    long long base = ((long long)blockIdx.y * S + m) * S;
    const float* row = Sc + base;
    unsigned short* prow = P + base;
    int tid = threadIdx.x;

    float mx = -3.4e38f;
    for (int n = tid; n <= m; n += 256) mx = fmaxf(mx, row[n]);
    red[tid] = mx; __syncthreads();
    for (int s = 128; s > 0; s >>= 1) {
        if (tid < s) red[tid] = fmaxf(red[tid], red[tid + s]);
        __syncthreads();
    }
    mx = red[0]; __syncthreads();

    float sum = 0.0f;
    for (int n = tid; n <= m; n += 256) sum += __expf(row[n] - mx);
    red[tid] = sum; __syncthreads();
    for (int s = 128; s > 0; s >>= 1) {
        if (tid < s) red[tid] += red[tid + s];
        __syncthreads();
    }
    float inv = 1.0f / red[0];

    for (int n = tid; n < S; n += 256) {
        float p = (n <= m) ? __expf(row[n] - mx) * inv : 0.0f;
        prow[n] = f32_to_bf16(p);
    }
}

// ---------------------------------------------------------------------------
// h = LayerNorm(h + att) * g + b ; writes f32 h and bf16 mirror
// ---------------------------------------------------------------------------
__global__ __launch_bounds__(256) void add_ln_kernel(
    float* __restrict__ h, unsigned short* __restrict__ hbf,
    const float* __restrict__ att,
    const float* __restrict__ g, const float* __restrict__ bta, int D)
{
    __shared__ float red[256];
    long long row = blockIdx.x;
    float* hr = h + row * D;
    const float* ar = att + row * D;
    unsigned short* hb = hbf + row * D;
    int tid = threadIdx.x;

    float s = 0.0f, s2 = 0.0f;
    for (int d = tid; d < D; d += 256) {
        float v = hr[d] + ar[d];
        s += v; s2 += v * v;
    }
    red[tid] = s; __syncthreads();
    for (int st = 128; st > 0; st >>= 1) {
        if (tid < st) red[tid] += red[tid + st];
        __syncthreads();
    }
    float mu = red[0] / D; __syncthreads();
    red[tid] = s2; __syncthreads();
    for (int st = 128; st > 0; st >>= 1) {
        if (tid < st) red[tid] += red[tid + st];
        __syncthreads();
    }
    float var = red[0] / D - mu * mu;
    float rstd = rsqrtf(var + 1e-5f);

    for (int d = tid; d < D; d += 256) {
        float v = (hr[d] + ar[d] - mu) * rstd * g[d] + bta[d];
        hr[d] = v;
        hb[d] = f32_to_bf16(v);
    }
}

// ---------------------------------------------------------------------------
// Host-side orchestration
// ---------------------------------------------------------------------------
extern "C" void kernel_launch(void* const* d_in, const int* in_sizes, int n_in,
                              void* d_out, int out_size, void* d_ws, size_t ws_size,
                              hipStream_t stream)
{
    (void)in_sizes; (void)n_in; (void)out_size; (void)ws_size;

    const int Bb = 2, S = 2048, D = 1024, DA = 512, DFF = 2048, V = 32000, L = 12;
    const long long MS = (long long)Bb * S;            // 4096 folded rows

    const int*   x    = (const int*)  d_in[0];
    const float* emb  = (const float*)d_in[1];
    const float* Wq   = (const float*)d_in[2];
    const float* bq   = (const float*)d_in[3];
    const float* Wk   = (const float*)d_in[4];
    const float* bk   = (const float*)d_in[5];
    const float* Wv   = (const float*)d_in[6];
    const float* bv   = (const float*)d_in[7];
    const float* lng  = (const float*)d_in[8];
    const float* lnb  = (const float*)d_in[9];
    const float* W1   = (const float*)d_in[10];
    const float* b1   = (const float*)d_in[11];
    const float* W2   = (const float*)d_in[12];
    const float* b2   = (const float*)d_in[13];
    const float* Wout = (const float*)d_in[14];
    const float* bout = (const float*)d_in[15];
    float* out = (float*)d_out;

    // ---- carve workspace (256B aligned) ----
    char* wp = (char*)d_ws;
    auto carve = [&](size_t bytes) -> void* {
        void* p = (void*)wp;
        wp += (bytes + 255) & ~(size_t)255;
        return p;
    };
    unsigned short* wq_t   = (unsigned short*)carve((size_t)L * DA * D * 2);
    unsigned short* wk_t   = (unsigned short*)carve((size_t)L * DA * D * 2);
    unsigned short* wv_t   = (unsigned short*)carve((size_t)L * D * D * 2);
    unsigned short* w1_t   = (unsigned short*)carve((size_t)L * DFF * D * 2);
    unsigned short* w2_t   = (unsigned short*)carve((size_t)L * D * DFF * 2);
    unsigned short* wout_t = (unsigned short*)carve((size_t)V * D * 2);
    float*          h      = (float*)         carve((size_t)MS * D * 4);
    unsigned short* hbf    = (unsigned short*)carve((size_t)MS * D * 2);
    unsigned short* qbf    = (unsigned short*)carve((size_t)MS * DA * 2);
    unsigned short* kbf    = (unsigned short*)carve((size_t)MS * DA * 2);
    unsigned short* vbf    = (unsigned short*)carve((size_t)MS * D * 2);
    float*          sc     = (float*)         carve((size_t)Bb * S * S * 4);
    unsigned short* pbf    = (unsigned short*)carve((size_t)Bb * S * S * 2);
    float*          att    = (float*)         carve((size_t)MS * D * 4);
    unsigned short* ffnbf  = (unsigned short*)carve((size_t)MS * DFF * 2);

    // ---- one-time (per call) weight convert+transpose to bf16 N-major ----
    convT_kernel<<<dim3(1024, L), 256, 0, stream>>>(Wq, wq_t, D, DA);
    convT_kernel<<<dim3(1024, L), 256, 0, stream>>>(Wk, wk_t, D, DA);
    convT_kernel<<<dim3(1024, L), 256, 0, stream>>>(Wv, wv_t, D, D);
    convT_kernel<<<dim3(2048, L), 256, 0, stream>>>(W1, w1_t, D, DFF);
    convT_kernel<<<dim3(2048, L), 256, 0, stream>>>(W2, w2_t, DFF, D);
    convT_kernel<<<dim3(8192, 1), 256, 0, stream>>>(Wout, wout_t, D, V);

    // ---- embedding gather ----
    embed_kernel<<<dim3((unsigned)MS), 256, 0, stream>>>(x, emb, h, hbf, D);

    for (int i = 0; i < L; ++i) {
        const unsigned short* wq_i = wq_t + (size_t)i * DA * D;
        const unsigned short* wk_i = wk_t + (size_t)i * DA * D;
        const unsigned short* wv_i = wv_t + (size_t)i * D * D;
        const unsigned short* w1_i = w1_t + (size_t)i * DFF * D;
        const unsigned short* w2_i = w2_t + (size_t)i * D * DFF;
        const float* bq_i = bq + (size_t)i * DA;
        const float* bk_i = bk + (size_t)i * DA;
        const float* bv_i = bv + (size_t)i * D;
        const float* b1_i = b1 + (size_t)i * DFF;
        const float* b2_i = b2 + (size_t)i * D;
        const float* g_i  = lng + (size_t)i * D;
        const float* be_i = lnb + (size_t)i * D;

        // q = h @ Wq + bq -> bf16 [MS, DA]
        gemm_bf16_kernel<true, false><<<dim3(DA / BN, MS / BM, 1), 256, 0, stream>>>(
            hbf, 0, D, wq_i, 0, D, bq_i,
            nullptr, 0, qbf, 0, (int)MS, DA, D, DA, 0);
        // k = h @ Wk + bk -> bf16 [MS, DA]
        gemm_bf16_kernel<true, false><<<dim3(DA / BN, MS / BM, 1), 256, 0, stream>>>(
            hbf, 0, D, wk_i, 0, D, bk_i,
            nullptr, 0, kbf, 0, (int)MS, DA, D, DA, 0);
        // v = h @ Wv + bv -> bf16 [MS, D]
        gemm_bf16_kernel<true, false><<<dim3(D / BN, MS / BM, 1), 256, 0, stream>>>(
            hbf, 0, D, wv_i, 0, D, bv_i,
            nullptr, 0, vbf, 0, (int)MS, D, D, D, 0);

        // scores = q @ k^T (per batch), causal-skipped -> f32 [B,S,S]
        gemm_bf16_kernel<true, false><<<dim3(S / BN, S / BM, Bb), 256, 0, stream>>>(
            qbf, (long long)S * DA, DA, kbf, (long long)S * DA, DA, nullptr,
            sc, (long long)S * S, nullptr, 0, S, S, DA, S, 1);

        // causal softmax -> bf16 P
        softmax_kernel<<<dim3(S, Bb), 256, 0, stream>>>(sc, pbf, S);

        // att = P @ v (per batch) -> f32 [B,S,D]
        gemm_bf16_kernel<false, false><<<dim3(D / BN, S / BM, Bb), 256, 0, stream>>>(
            pbf, (long long)S * S, S, vbf, (long long)S * D, D, nullptr,
            att, (long long)S * D, nullptr, 0, S, D, S, D, 0);

        // h = LN(h + att) * g + b   (f32 + bf16 mirror)
        add_ln_kernel<<<dim3((unsigned)MS), 256, 0, stream>>>(h, hbf, att, g_i, be_i, D);

        // ffn1 = relu(h @ W1 + b1) -> bf16 [MS, DFF]
        gemm_bf16_kernel<true, true><<<dim3(DFF / BN, MS / BM, 1), 256, 0, stream>>>(
            hbf, 0, D, w1_i, 0, D, b1_i,
            nullptr, 0, ffnbf, 0, (int)MS, DFF, D, DFF, 0);
        // h = ffn1 @ W2 + b2 (replaces h; no residual, faithful)
        gemm_bf16_kernel<true, false><<<dim3(D / BN, MS / BM, 1), 256, 0, stream>>>(
            ffnbf, 0, DFF, w2_i, 0, DFF, b2_i,
            h, 0, hbf, 0, (int)MS, D, DFF, D, 0);
    }

    // logits = h @ Wout + bout -> f32 [B,S,V] (d_out)
    gemm_bf16_kernel<true, false><<<dim3(V / BN, MS / BM, 1), 256, 0, stream>>>(
        hbf, 0, D, wout_t, 0, D, bout,
        out, 0, nullptr, 0, (int)MS, V, D, V, 0);
}
